// OutputGlobalModel_5995774345766
// MI455X (gfx1250) — matrix operand order repaired and checked by reading, not compiled, not run on hardware
//
#include <hip/hip_runtime.h>

// ---------------------------------------------------------------------------
// OutputGlobalModel for MI455X (gfx1250, wave32)
//   Phase 1: bucket edges by graph (2M int atomics total), then atomic-free
//            gather-max pooling: streams the 1.02 GB edge tensor at HBM rate
//            (~44 us @ 23.3 TB/s) with coalesced 1KB-row reads. Avoids the
//            256M float-max atomics of the naive scatter (L2 RMW-rate bound).
//   Phase 2: 3-layer MLP head via v_wmma_f32_16x16x32_f16 (f32 accumulate),
//            16x64 output block per wave, guard-free inner loop (padded Wout)
// ---------------------------------------------------------------------------

typedef __attribute__((ext_vector_type(8)))  _Float16 v8h;
typedef __attribute__((ext_vector_type(16))) _Float16 v16h;
typedef __attribute__((ext_vector_type(8)))  float    v8f;

#define E_EDGES  1000000
#define NUM_G    4096
#define IN_F     256
#define FC_F     512
#define N_CLS    100
#define N_CLS_P  128          // Wout rows padded with zeros -> guard-free K loop
#define FILL_TH  (-10000.0f)

// ---------------- tiny int zero-fill ---------------------------------------
__global__ void zero_i32_kernel(int* __restrict__ p, int n) {
  int i = blockIdx.x * blockDim.x + threadIdx.x;
  if (i < n) p[i] = 0;
}

// ------- f32 -> f16 weight conversion with zero-padding to Npad rows --------
__global__ void convert_pad_kernel(const float* __restrict__ src,
                                   _Float16* __restrict__ dst,
                                   int N, int K, int Npad) {
  int i = blockIdx.x * blockDim.x + threadIdx.x;
  if (i < Npad * K) {
    int row = i / K;
    dst[i] = (row < N) ? (_Float16)src[i] : (_Float16)0.0f;
  }
}

// ---------------- Phase 1a: gather graph id per edge + histogram ------------
__global__ void count_kernel(const int* __restrict__ edge_src,  // row 0 of edge_indices
                             const int* __restrict__ n2g,
                             int* __restrict__ edge_g,
                             int* __restrict__ counts) {
  int e = blockIdx.x * blockDim.x + threadIdx.x;
  if (e < E_EDGES) {
    int g = n2g[edge_src[e]];
    edge_g[e] = g;
    atomicAdd(&counts[g], 1);
  }
}

// ---------------- Phase 1b: exclusive scan of 4096 counters (1 block) -------
__global__ __launch_bounds__(256)
void scan_kernel(const int* __restrict__ counts,
                 int* __restrict__ offsets,    // [NUM_G+1]
                 int* __restrict__ cursors) {  // [NUM_G]
  __shared__ int partial[256];
  int t = threadIdx.x;
  int base = t * 16;                           // 256 threads x 16 = 4096
  int local[16];
  int s = 0;
#pragma unroll
  for (int i = 0; i < 16; ++i) { local[i] = s; s += counts[base + i]; }
  partial[t] = s;
  __syncthreads();
  for (int off = 1; off < 256; off <<= 1) {    // Hillis-Steele inclusive scan
    int v = (t >= off) ? partial[t - off] : 0;
    __syncthreads();
    partial[t] += v;
    __syncthreads();
  }
  int chunkBase = (t == 0) ? 0 : partial[t - 1];
#pragma unroll
  for (int i = 0; i < 16; ++i) {
    int off = chunkBase + local[i];
    offsets[base + i] = off;
    cursors[base + i] = off;
  }
  if (t == 255) offsets[NUM_G] = partial[255];
}

// ---------------- Phase 1c: scatter edge ids into per-graph buckets ---------
__global__ void bucket_kernel(const int* __restrict__ edge_g,
                              int* __restrict__ cursors,
                              int* __restrict__ bucket) {
  int e = blockIdx.x * blockDim.x + threadIdx.x;
  if (e < E_EDGES) {
    int g = edge_g[e];
    int pos = atomicAdd(&cursors[g], 1);       // bucket order irrelevant: max is commutative
    bucket[pos] = e;
  }
}

// ---------------- Phase 1d: atomic-free gather-max + fixup + f16 ------------
// One 256-thread block per graph; thread = feature column. Each edge row read
// is a fully coalesced 1KB burst; edge rows are touched exactly once overall.
__global__ __launch_bounds__(256)
void pool_max_kernel(const float* __restrict__ edges,
                     const int*   __restrict__ bucket,
                     const int*   __restrict__ offsets,
                     _Float16*    __restrict__ a0) {     // [NUM_G, IN_F] f16
  int g = blockIdx.x;
  int t = threadIdx.x;                          // feature index 0..255
  int start = offsets[g];
  int end   = offsets[g + 1];
  float acc = -__builtin_inff();
  int i = start;
  for (; i + 4 <= end; i += 4) {                // 4-deep to keep loads in flight
    int e0 = bucket[i + 0], e1 = bucket[i + 1];
    int e2 = bucket[i + 2], e3 = bucket[i + 3];
    float v0 = edges[(size_t)e0 * IN_F + t];
    float v1 = edges[(size_t)e1 * IN_F + t];
    float v2 = edges[(size_t)e2 * IN_F + t];
    float v3 = edges[(size_t)e3 * IN_F + t];
    acc = fmaxf(acc, fmaxf(fmaxf(v0, v1), fmaxf(v2, v3)));
  }
  for (; i < end; ++i)
    acc = fmaxf(acc, edges[(size_t)bucket[i] * IN_F + t]);
  acc = (acc < FILL_TH) ? 0.0f : acc;           // empty segment -> -inf -> 0
  a0[(size_t)g * IN_F + t] = (_Float16)acc;
}

// ---------------- Phase 2: WMMA GEMM  out = relu?(A @ W^T + bias) -----------
// A: [M,K] f16 row-major. Wp: [Npad,K] f16 row-major, Npad = 64*tilesN64
// (zero-padded) -> guard-free K loop. One wave computes a 16x64 output block
// (4 accumulators); the A fragment is reused 4x per K step.
template <bool RELU, bool OUT16>
__global__ __launch_bounds__(256)
void gemm_wmma_kernel(const _Float16* __restrict__ A,
                      const _Float16* __restrict__ Wp,
                      const float*    __restrict__ bias,   // [N]
                      void* __restrict__ outv,             // [M,N]
                      int M, int N, int K, int tilesN64) {
  int wave = (int)((blockIdx.x * blockDim.x + threadIdx.x) >> 5);
  unsigned lane = threadIdx.x & 31;
  int tilesM = M >> 4;
  if (wave >= tilesM * tilesN64) return;        // wave-uniform

  int tn = wave % tilesN64;
  int tm = wave / tilesN64;
  int mBase = tm << 4;
  int nBase = tn << 6;                          // 64-wide N block

  int mLocal = (int)(lane & 15);
  int half   = (int)(lane >> 4);                // 0 or 1

  // A lane layout (16-bit A 16x32): row = lane&15; K = 8*half..+7, 16+8*half..+7
  const _Float16* arow = A + (size_t)(mBase + mLocal) * K + half * 8;
  // B lane layout (B 32x16): col(lane) = lane&15 -> W row; K = 8*half..+7, 16+8*half..+7
  const _Float16* wbase = Wp + (size_t)(nBase + mLocal) * K + half * 8;

  v8f acc[4] = {{}, {}, {}, {}};
  for (int k0 = 0; k0 < K; k0 += 32) {
    v16h a;
    {
      v8h alo = *(const v8h*)(arow + k0);
      v8h ahi = *(const v8h*)(arow + k0 + 16);
#pragma unroll
      for (int i = 0; i < 8; ++i) { a[i] = alo[i]; a[8 + i] = ahi[i]; }
    }
#pragma unroll
    for (int c = 0; c < 4; ++c) {
      const _Float16* wrow = wbase + (size_t)(c * 16) * K;
      v16h b;
      v8h blo = *(const v8h*)(wrow + k0);
      v8h bhi = *(const v8h*)(wrow + k0 + 16);
#pragma unroll
      for (int i = 0; i < 8; ++i) { b[i] = blo[i]; b[8 + i] = bhi[i]; }
      acc[c] = __builtin_amdgcn_wmma_f32_16x16x32_f16(
          /*neg_a=*/false, a, /*neg_b=*/false, b,
          /*c_mod=*/(short)0, acc[c], /*reuse_a=*/false, /*reuse_b=*/false);
    }
  }

#pragma unroll
  for (int c = 0; c < 4; ++c) {
    int n = nBase + c * 16 + mLocal;            // output column of this lane
    float bv = (n < N) ? bias[n] : 0.0f;
#pragma unroll
    for (int v = 0; v < 8; ++v) {               // C/D: VGPR v -> row v + 8*half
      int row = mBase + v + half * 8;
      float val = acc[c][v] + bv;
      if (RELU) val = fmaxf(val, 0.0f);
      if (n < N) {
        if (OUT16) ((_Float16*)outv)[(size_t)row * N + n] = (_Float16)val;
        else       ((float*)   outv)[(size_t)row * N + n] = val;
      }
    }
  }
}

// ---------------------------------------------------------------------------
extern "C" void kernel_launch(void* const* d_in, const int* in_sizes, int n_in,
                              void* d_out, int out_size, void* d_ws, size_t ws_size,
                              hipStream_t stream) {
  const float* edges = (const float*)d_in[0];   // [E, 256]
  const int*   eidx  = (const int*)  d_in[1];   // [2, E]; row 0 = src nodes
  const int*   n2g   = (const int*)  d_in[2];   // [N_NODES]
  // d_in[3] = num_graphs scalar (compile-time constant NUM_G)
  const float* W0   = (const float*)d_in[4];    // [512,256]
  const float* b0   = (const float*)d_in[5];
  const float* W1   = (const float*)d_in[6];    // [512,512]
  const float* b1   = (const float*)d_in[7];
  const float* Wout = (const float*)d_in[8];    // [100,512]
  const float* bout = (const float*)d_in[9];
  float* out = (float*)d_out;                   // [4096,100]

  // ---- workspace carve (~19 MB) ----
  char* p = (char*)d_ws;
  _Float16* A0      = (_Float16*)p; p += (size_t)NUM_G   * IN_F * 2;
  _Float16* H0      = (_Float16*)p; p += (size_t)NUM_G   * FC_F * 2;
  _Float16* H1      = (_Float16*)p; p += (size_t)NUM_G   * FC_F * 2;
  _Float16* W0h     = (_Float16*)p; p += (size_t)FC_F    * IN_F * 2;
  _Float16* W1h     = (_Float16*)p; p += (size_t)FC_F    * FC_F * 2;
  _Float16* Wouth   = (_Float16*)p; p += (size_t)N_CLS_P * FC_F * 2;  // zero-padded
  int* edge_g       = (int*)p;      p += (size_t)E_EDGES * 4;
  int* bucket       = (int*)p;      p += (size_t)E_EDGES * 4;
  int* counts       = (int*)p;      p += (size_t)NUM_G * 4;
  int* offsets      = (int*)p;      p += (size_t)(NUM_G + 1) * 4;
  int* cursors      = (int*)p;      p += (size_t)NUM_G * 4;

  const int T = 256;
  const int EB = (E_EDGES + T - 1) / T;

  // Phase 0: weight conversion (Wout padded to 128 zero rows)
  convert_pad_kernel<<<(FC_F * IN_F + T - 1) / T, T, 0, stream>>>(W0, W0h, FC_F, IN_F, FC_F);
  convert_pad_kernel<<<(FC_F * FC_F + T - 1) / T, T, 0, stream>>>(W1, W1h, FC_F, FC_F, FC_F);
  convert_pad_kernel<<<(N_CLS_P * FC_F + T - 1) / T, T, 0, stream>>>(Wout, Wouth, N_CLS, FC_F, N_CLS_P);

  // Phase 1: bucket edges by graph, then atomic-free gather-max pooling
  zero_i32_kernel<<<(NUM_G + T - 1) / T, T, 0, stream>>>(counts, NUM_G);
  count_kernel <<<EB, T, 0, stream>>>(eidx, n2g, edge_g, counts);
  scan_kernel  <<<1,  T, 0, stream>>>(counts, offsets, cursors);
  bucket_kernel<<<EB, T, 0, stream>>>(edge_g, cursors, bucket);
  pool_max_kernel<<<NUM_G, T, 0, stream>>>(edges, bucket, offsets, A0);

  // Phase 2: MLP head. Waves per GEMM: 256*8, 256*8, 256*2 (16x64 block/wave)
  gemm_wmma_kernel<true,  true ><<<(256 * 8) * 32 / T, T, 0, stream>>>(A0, W0h, b0, H0, NUM_G, FC_F, IN_F, FC_F / 64);
  gemm_wmma_kernel<true,  true ><<<(256 * 8) * 32 / T, T, 0, stream>>>(H0, W1h, b1, H1, NUM_G, FC_F, FC_F, FC_F / 64);
  gemm_wmma_kernel<false, false><<<(256 * 2) * 32 / T, T, 0, stream>>>(H1, Wouth, bout, out, NUM_G, N_CLS, FC_F, N_CLS_P / 64);
}
